// VariationalQuanvolutionCircuit_65481071396637
// MI455X (gfx1250) — matrix operand order repaired and checked by reading, not compiled, not run on hardware
//
#include <hip/hip_runtime.h>

typedef __attribute__((ext_vector_type(16))) _Float16 v16h;
typedef __attribute__((ext_vector_type(8)))  float    v8f;

// ---------- complex helpers ----------
__device__ __forceinline__ float2 cmul(float2 a, float2 b) {
    return make_float2(a.x * b.x - a.y * b.y, a.x * b.y + a.y * b.x);
}
__device__ __forceinline__ float2 cadd(float2 a, float2 b) {
    return make_float2(a.x + b.x, a.y + b.y);
}

// =====================================================================
// Setup kernel (1 block, 256 threads): build the patch-independent
// 16x16 circuit unitary V from params, fold in the (-i)^popcount(k)
// phases of the RX product state, and emit real matrices P,Q split into
// f16 hi/lo pairs (row-major [n][k]) into d_ws:
//   ws[0..255]=P_hi, ws[256..511]=P_lo, ws[512..767]=Q_hi, ws[768..1023]=Q_lo
// =====================================================================
__global__ void vq_setup(const float* __restrict__ params, _Float16* __restrict__ ws) {
    __shared__ float2 V[256];              // V[row*16 + col]
    const int t   = threadIdx.x;           // 0..255
    const int row = t >> 4;
    const int col = t & 15;

    V[t] = make_float2(row == col ? 1.0f : 0.0f, 0.0f);
    __syncthreads();

    for (int l = 0; l < 3; ++l) {
        // Rot(phi,theta,omega) = RZ(omega) RY(theta) RZ(phi) on each qubit
        for (int q = 0; q < 4; ++q) {
            const float phi   = params[(l * 4 + q) * 3 + 0];
            const float theta = params[(l * 4 + q) * 3 + 1];
            const float omega = params[(l * 4 + q) * 3 + 2];
            float sth, cth; __sincosf(0.5f * theta,         &sth, &cth);
            float sap, cap; __sincosf(0.5f * (phi + omega), &sap, &cap);
            float sam, cam; __sincosf(0.5f * (phi - omega), &sam, &cam);
            const float2 U00 = make_float2( cap * cth, -sap * cth);  //  e^{-i(phi+omega)/2} cos
            const float2 U01 = make_float2(-cam * sth, -sam * sth);  // -e^{+i(phi-omega)/2} sin
            const float2 U10 = make_float2( cam * sth, -sam * sth);  //  e^{-i(phi-omega)/2} sin
            const float2 U11 = make_float2( cap * cth,  sap * cth);  //  e^{+i(phi+omega)/2} cos

            const int mask = 1 << (3 - q);                 // qubit q <-> bit (3-q)
            const float2 vs = V[t];
            const float2 vp = V[(row ^ mask) * 16 + col];
            __syncthreads();
            float2 nv;
            if (row & mask) nv = cadd(cmul(U10, vp), cmul(U11, vs));
            else            nv = cadd(cmul(U00, vs), cmul(U01, vp));
            V[t] = nv;
            __syncthreads();
        }
        // CNOT chain: control q, target q+1  => row permutation (left-multiply)
        for (int q = 0; q < 3; ++q) {
            const int cm = 1 << (3 - q);
            const int tm = 1 << (2 - q);
            const int srow = (row & cm) ? (row ^ tm) : row;
            const float2 tmp = V[srow * 16 + col];
            __syncthreads();
            V[t] = tmp;
            __syncthreads();
        }
    }

    // Fold (-i)^popcount(col):  W = V * diag((-i)^pc);  P = Re W, Q = Im W
    const float2 v = V[t];
    const int pc = __popc((unsigned)col) & 3;
    float P, Q;
    switch (pc) {
        case 0:  P =  v.x; Q =  v.y; break;
        case 1:  P =  v.y; Q = -v.x; break;   // *(-i)
        case 2:  P = -v.x; Q = -v.y; break;   // *(-1)
        default: P = -v.y; Q =  v.x; break;   // *(+i)
    }
    const _Float16 Ph = (_Float16)P;
    const _Float16 Qh = (_Float16)Q;
    ws[t]       = Ph;
    ws[256 + t] = (_Float16)(P - (float)Ph);
    ws[512 + t] = Qh;
    ws[768 + t] = (_Float16)(Q - (float)Qh);
}

// =====================================================================
// Main kernel: one wave32 per 16-patch tile iteration.
//   D_P = R*P^T, D_Q = R*Q^T via compensated f16 WMMA (hi/lo split,
//   3x v_wmma_f32_16x16x32_f16 each), prob = D_P^2 + D_Q^2, then an LDS
//   transpose + butterfly signed sums + one cross-half shfl for <Z_q>.
// =====================================================================
__global__ __launch_bounds__(256) void vq_main(const float4* __restrict__ x,
                                               const _Float16* __restrict__ ws,
                                               float* __restrict__ out, int ntiles) {
    // per-wave transpose scratch: 16 rows * stride 20 floats (conflict-free)
    __shared__ __align__(16) float lds[8 * 320];
    const int lane = threadIdx.x & 31;
    const int n    = lane & 15;             // N index / patch-in-tile for r
    const bool hiw = lane >= 16;
    float* wlds = lds + (threadIdx.x >> 5) * 320;

    // B operands (constant per launch): B[k][nc] = P[nc][k]; lane half h holds
    // K rows 8h..8h+7 in elements 0..7 (real data), elements 8..15 zero pad.
    v16h bPh = {}, bPl = {}, bQh = {}, bQl = {};
    {
        const _Float16* w = ws + n * 16 + (hiw ? 8 : 0);
#pragma unroll
        for (int j = 0; j < 8; ++j) {
            bPh[j] = w[j];  bPl[j] = w[256 + j];
            bQh[j] = w[512 + j];  bQl[j] = w[768 + j];
        }
    }

    const int wid = blockIdx.x * (blockDim.x >> 5) + (threadIdx.x >> 5);
    const int nw  = gridDim.x * (blockDim.x >> 5);

    for (int tile = wid; tile < ntiles; tile += nw) {
        // ---- r values for patch (tile*16 + n); lane covers K = 8*hiw + j ----
        const float4 xv = x[(size_t)tile * 16 + n];
        float s0, c0, s1, c1, s2, c2, s3, c3;
        __sincosf(0.5f * xv.x, &s0, &c0);   // qubit0 <-> bit 8
        __sincosf(0.5f * xv.y, &s1, &c1);   // qubit1 <-> bit 4
        __sincosf(0.5f * xv.z, &s2, &c2);   // qubit2 <-> bit 2
        __sincosf(0.5f * xv.w, &s3, &c3);   // qubit3 <-> bit 1
        const float ca = hiw ? s0 : c0;     // bit3 of K == hiw
        const float ta = ca * c1, tb = ca * s1;
        const float u0 = c2 * c3, u1 = c2 * s3, u2 = s2 * c3, u3 = s2 * s3;
        float r8[8] = { ta * u0, ta * u1, ta * u2, ta * u3,
                        tb * u0, tb * u1, tb * u2, tb * u3 };

        // A operand hi/lo split: lo lanes K=0..7, hi lanes K=8..15, rest zero
        v16h ah = {}, al = {};
#pragma unroll
        for (int j = 0; j < 8; ++j) {
            const _Float16 h = (_Float16)r8[j];
            ah[j] = h;
            al[j] = (_Float16)(r8[j] - (float)h);
        }

        // ---- compensated GEMMs (small terms first) ----
        v8f accP = {}, accQ = {};
        accP = __builtin_amdgcn_wmma_f32_16x16x32_f16(false, al, false, bPh, (short)0, accP, false, false);
        accP = __builtin_amdgcn_wmma_f32_16x16x32_f16(false, ah, false, bPl, (short)0, accP, false, false);
        accP = __builtin_amdgcn_wmma_f32_16x16x32_f16(false, ah, false, bPh, (short)0, accP, false, false);
        accQ = __builtin_amdgcn_wmma_f32_16x16x32_f16(false, al, false, bQh, (short)0, accQ, false, false);
        accQ = __builtin_amdgcn_wmma_f32_16x16x32_f16(false, ah, false, bQl, (short)0, accQ, false, false);
        accQ = __builtin_amdgcn_wmma_f32_16x16x32_f16(false, ah, false, bQh, (short)0, accQ, false, false);

        // ---- prob -> LDS transpose (C layout: lane holds [m=v+8*hiw][n]) ----
#pragma unroll
        for (int v = 0; v < 8; ++v) {
            const float p = accP[v] * accP[v] + accQ[v] * accQ[v];
            wlds[(v + (hiw ? 8 : 0)) * 20 + n] = p;
        }
        asm volatile("s_wait_dscnt 0x0" ::: "memory");

        // lane reads row m = n, columns 8*hiw .. 8*hiw+7 (two aligned float4)
        const float* rp = wlds + n * 20 + (hiw ? 8 : 0);
        const float4 pA = *(const float4*)(rp);
        const float4 pB = *(const float4*)(rp + 4);
        const float p0 = pA.x, p1 = pA.y, p2 = pA.z, p3 = pA.w;
        const float p4 = pB.x, p5 = pB.y, p6 = pB.z, p7 = pB.w;

        // signed sums over this lane's 8 columns (sign of bit3 is wave-half
        // uniform; bits 2..0 are compile-time per element)
        const float s01 = p0 + p1, s23 = p2 + p3, s45 = p4 + p5, s67 = p6 + p7;
        const float d01 = p0 - p1, d23 = p2 - p3, d45 = p4 - p5, d67 = p6 - p7;
        const float sA = s01 + s23, sB = s45 + s67;
        const float T  = sA + sB;
        float z0 = hiw ? -T : T;                       // qubit0: sign = bit3
        float z1 = sA - sB;                            // qubit1: sign = bit2
        float z2 = (s01 - s23) + (s45 - s67);          // qubit2: sign = bit1
        float z3 = (d01 + d23) + (d45 + d67);          // qubit3: sign = bit0
        z0 += __shfl_xor(z0, 16, 32);                  // merge column halves
        z1 += __shfl_xor(z1, 16, 32);
        z2 += __shfl_xor(z2, 16, 32);
        z3 += __shfl_xor(z3, 16, 32);

        // out[patch][q]: lane n (lo half) stores patch = tile*16+n as float4
        if (!hiw) {
            ((float4*)out)[(size_t)tile * 16 + n] = make_float4(z0, z1, z2, z3);
        }
    }
}

extern "C" void kernel_launch(void* const* d_in, const int* in_sizes, int n_in,
                              void* d_out, int out_size, void* d_ws, size_t ws_size,
                              hipStream_t stream) {
    const float*  x      = (const float*)d_in[0];   // (256,4096,4) f32
    const float*  params = (const float*)d_in[1];   // (3,4,3) f32
    float*        out    = (float*)d_out;           // (256,4096,4) f32
    _Float16*     ws     = (_Float16*)d_ws;         // 2 KB used: P/Q hi+lo f16

    const int M      = in_sizes[0] / 4;             // patches
    const int ntiles = M / 16;

    vq_setup<<<1, 256, 0, stream>>>(params, ws);
    vq_main<<<2048, 256, 0, stream>>>((const float4*)x, ws, out, ntiles);
}
